// FullTensorSquareSelfInteraction_42279658062138
// MI455X (gfx1250) — matrix operand order repaired and checked by reading, not compiled, not run on hardware
//
#include <hip/hip_runtime.h>
#include <math.h>

// ---------------- problem constants (match reference) ----------------
#define NSAMP 8192
#define C0N   64
#define C1N   32
#define P0N   2080      // C0*(C0+1)/2
#define P1N   528       // C1*(C1+1)/2
#define A1N   496       // C1*(C1-1)/2
#define M0N   2672      // C0 + P0 + P1
#define M1N   2080      // C1 + C0*C1
#define NIRR  5776      // M0 + M1 + A1 + P1
#define KP    2688      // padded M0 (multiple of 128)
#define GP    5888      // padded NIRR (multiple of 128)
#define OSTR  288       // 64 + 96 + 48 + 80

typedef unsigned short u16;
typedef __attribute__((ext_vector_type(16))) __bf16 bf16x16;
typedef __attribute__((ext_vector_type(8)))  __bf16 bf16x8;
typedef __attribute__((ext_vector_type(8)))  float  f32x8;

union BFU { __bf16 b; u16 u; };
static __device__ inline __bf16 u2b(u16 u){ BFU x; x.u=u; return x.b; }
static __device__ inline u16 f2bfu(float f){
  union {float f; unsigned u;} x; x.f=f;
  unsigned r = x.u + 0x7FFFu + ((x.u>>16)&1u);
  return (u16)(r>>16);
}
static __device__ inline float bfu2f(u16 u){
  union {unsigned u; float f;} x; x.u=((unsigned)u)<<16; return x.f;
}

// ---------------- pair index tables ----------------
__global__ void init_tables(int* t0, int* t1, int* tA) {
  int t = threadIdx.x;
  if (t == 0) { int p = 0; for (int a = 0; a < C0N; ++a) for (int b = a; b < C0N; ++b) t0[p++] = (a<<8)|b; }
  else if (t == 1) { int p = 0; for (int a = 0; a < C1N; ++a) for (int b = a; b < C1N; ++b) t1[p++] = (a<<8)|b; }
  else if (t == 2) { int p = 0; for (int a = 0; a < C1N; ++a) for (int b = a+1; b < C1N; ++b) tA[p++] = (a<<8)|b; }
}

// ---------------- f32 weight -> zero-padded bf16 ----------------
__global__ void cvt_pad(const float* __restrict__ src, u16* __restrict__ dst,
                        int srcR, int srcC, int dstC, long total) {
  long idx = (long)blockIdx.x * blockDim.x + threadIdx.x;
  if (idx >= total) return;
  int r = (int)(idx / dstC), c = (int)(idx - (long)r * dstC);
  float v = (r < srcR && c < srcC) ? src[(size_t)r * srcC + c] : 0.0f;
  dst[idx] = f2bfu(v);
}

// ---------------- build f0 = [x_s | ss | vv0] as bf16, pitch KP ----------------
__global__ __launch_bounds__(256) void build_f0(const float* __restrict__ x_s,
                                                const float* __restrict__ x_v,
                                                const int* __restrict__ t0,
                                                const int* __restrict__ t1,
                                                u16* __restrict__ f0b) {
  __shared__ float xs[C0N];
  __shared__ float xv[C1N * 3];
  int n = blockIdx.x, tid = threadIdx.x;
  if (tid < C0N) xs[tid] = x_s[(size_t)n * C0N + tid];
  if (tid >= 64 && tid < 64 + C1N * 3) xv[tid - 64] = x_v[(size_t)n * C1N * 3 + tid - 64];
  __syncthreads();
  const float s2 = 0.70710678118654752f;
  const float i3 = 0.57735026918962576f;   // 1/sqrt(3)
  u16* row = f0b + (size_t)n * KP;
  for (int col = tid; col < KP; col += 256) {
    float v;
    if (col < C0N) {
      v = xs[col];
    } else if (col < C0N + P0N) {
      int p = col - C0N, ab = t0[p], a = ab >> 8, b = ab & 255;
      v = xs[a] * xs[b] * (a == b ? s2 : 1.0f);
    } else if (col < M0N) {
      int q = col - C0N - P0N, ab = t1[q], a = ab >> 8, b = ab & 255;
      float d = xv[a*3]*xv[b*3] + xv[a*3+1]*xv[b*3+1] + xv[a*3+2]*xv[b*3+2];
      v = d * (a == b ? s2 : 1.0f) * i3;
    } else {
      v = 0.0f;                            // zero pad
    }
    row[col] = f2bfu(v);
  }
}

// ---------------- WMMA bf16 GEMM: C(8192 x ldc) = A(8192 x K, pitch K) * B(K x ldb) ----------------
// Double-buffered; A tile staged via async global->LDS DMA (ASYNCcnt), B tile via
// VGPR round-trip with K-transpose scatter (latency hidden under WMMA compute).
// MODE 0: store f32   MODE 1: silu -> bf16   MODE 2: bf16
union Frag16 { bf16x16 v; bf16x8 h[2]; };

template<int MODE>
__global__ __launch_bounds__(256) void gemm_bf16_wmma(const u16* __restrict__ A,
                                                      const u16* __restrict__ B,
                                                      void* __restrict__ Cout,
                                                      int K, int ldb, int ldc) {
  __shared__ __bf16 As[2][128][40];   // row-major A tile (128 x 32) + pad, double buffered
  __shared__ __bf16 Bs[2][128][40];   // K-transposed B tile Bs[n][k], double buffered
  const int tid  = threadIdx.x;
  const int lane = tid & 31, wave = tid >> 5;
  const int wR = wave >> 2, wC = wave & 3;     // 2 x 4 wave grid, wave tile 64 x 32
  const int hl = lane >> 4, lr = lane & 15;
  const int rowBlock = blockIdx.y * 128;
  const int colBlock = blockIdx.x * 128;

  // A staging: 2 async b128 per thread (each lane moves 16B straight to LDS)
  const int ar0 = tid >> 2, akc = (tid & 3) * 8;     // rows 0..63
  const int ar1 = ar0 + 64;                          // rows 64..127
  // B staging: 2 b128 per thread, transposed scatter into LDS
  const int bk0 = tid >> 4, bn = (tid & 15) * 8;     // k 0..15
  const int bk1 = bk0 + 16;                          // k 16..31

  f32x8 acc[4][2] = {};
  uint4 bv0, bv1;

  auto asyncA = [&](int buf, int k0) {
    const u16* g0 = A + (size_t)(rowBlock + ar0) * K + k0 + akc;
    const u16* g1 = A + (size_t)(rowBlock + ar1) * K + k0 + akc;
    unsigned l0 = (unsigned)(size_t)&As[buf][ar0][akc];
    unsigned l1 = (unsigned)(size_t)&As[buf][ar1][akc];
    asm volatile("global_load_async_to_lds_b128 %0, %1, off\n\t"
                 "global_load_async_to_lds_b128 %2, %3, off"
                 :: "v"(l0), "v"(g0), "v"(l1), "v"(g1) : "memory");
  };
  auto loadBregs = [&](int k0) {
    bv0 = *(const uint4*)(B + (size_t)(k0 + bk0) * ldb + colBlock + bn);
    bv1 = *(const uint4*)(B + (size_t)(k0 + bk1) * ldb + colBlock + bn);
  };
  auto storeB = [&](int buf) {
    const u16* p0 = (const u16*)&bv0;
    const u16* p1 = (const u16*)&bv1;
    #pragma unroll
    for (int j = 0; j < 8; ++j) Bs[buf][bn + j][bk0] = u2b(p0[j]);
    #pragma unroll
    for (int j = 0; j < 8; ++j) Bs[buf][bn + j][bk1] = u2b(p1[j]);
  };

  // prologue: fill buffer 0
  asyncA(0, 0);
  loadBregs(0);
  storeB(0);
  asm volatile("s_wait_asynccnt 0x0" ::: "memory");
  __syncthreads();

  const int KT = K / 32;
  for (int kt = 0; kt < KT; ++kt) {
    const int cur = kt & 1, nxt = cur ^ 1;
    const bool more = (kt + 1 < KT);
    if (more) {
      asyncA(nxt, (kt + 1) * 32);   // DMA next A tile while we compute
      loadBregs((kt + 1) * 32);     // next B tile in flight in VGPRs
    }

    Frag16 af[4];
    #pragma unroll
    for (int fm = 0; fm < 4; ++fm) {
      const __bf16* src = &As[cur][wR*64 + fm*16 + lr][hl*8];  // lanes<16: K0-7,16-23 ; lanes>=16: K8-15,24-31
      af[fm].h[0] = *(const bf16x8*)(src);
      af[fm].h[1] = *(const bf16x8*)(src + 16);
    }
    Frag16 bfg[2];
    #pragma unroll
    for (int fn = 0; fn < 2; ++fn) {
      const __bf16* src = &Bs[cur][wC*32 + fn*16 + lr][hl*8];  // lane = column; lanes<16: K0-7,16-23 ; lanes>=16: K8-15,24-31
      bfg[fn].h[0] = *(const bf16x8*)(src);
      bfg[fn].h[1] = *(const bf16x8*)(src + 16);
    }
    #pragma unroll
    for (int fm = 0; fm < 4; ++fm)
      #pragma unroll
      for (int fn = 0; fn < 2; ++fn)
        acc[fm][fn] = __builtin_amdgcn_wmma_f32_16x16x32_bf16(
            false, af[fm].v, false, bfg[fn].v, (short)0, acc[fm][fn], false, false);

    if (more) {
      storeB(nxt);                                       // scatter next B tile
      asm volatile("s_wait_asynccnt 0x0" ::: "memory");  // next A tile resident
    }
    __syncthreads();
  }

  // epilogue: lane<16 -> rows r, N=lr ; lane>=16 -> rows 8+r
  #pragma unroll
  for (int fm = 0; fm < 4; ++fm) {
    int rowStart = rowBlock + wR*64 + fm*16 + hl*8;
    #pragma unroll
    for (int fn = 0; fn < 2; ++fn) {
      int col = colBlock + wC*32 + fn*16 + lr;
      #pragma unroll
      for (int r = 0; r < 8; ++r) {
        float v = acc[fm][fn][r];
        size_t idx = (size_t)(rowStart + r) * ldc + col;
        if (MODE == 0) {
          ((float*)Cout)[idx] = v;
        } else if (MODE == 1) {
          float s = v / (1.0f + __expf(-v));                // silu
          ((u16*)Cout)[idx] = f2bfu(s);
        } else {
          ((u16*)Cout)[idx] = f2bfu(v);
        }
      }
    }
  }
}

// ---------------- gate: f0b *= g0 (in place, bf16) ----------------
__global__ void gate_f0(u16* __restrict__ f0b, const u16* __restrict__ gb) {
  long idx = (long)blockIdx.x * blockDim.x + threadIdx.x;
  if (idx >= (long)NSAMP * KP) return;
  int n = (int)(idx / KP), k = (int)(idx - (long)n * KP);
  float v = bfu2f(f0b[idx]) * bfu2f(gb[(size_t)n * GP + k]);
  f0b[idx] = f2bfu(v);
}

// ---------------- per-sample small contractions + o0 copy ----------------
__global__ __launch_bounds__(256) void outputs_kernel(const float* __restrict__ x_s,
                                                      const float* __restrict__ x_v,
                                                      const u16* __restrict__ gb,
                                                      const float* __restrict__ o0p,
                                                      const float* __restrict__ w1o,
                                                      const float* __restrict__ w1e,
                                                      const float* __restrict__ w2e,
                                                      const int* __restrict__ tA,
                                                      const int* __restrict__ t1,
                                                      float* __restrict__ out) {
  __shared__ float xs[C0N];
  __shared__ float xv[C1N * 3];
  __shared__ float sve[A1N * 3];   // vv1e
  __shared__ float sv2[P1N * 5];   // vv2
  int n = blockIdx.x, tid = threadIdx.x;
  if (tid < C0N) xs[tid] = x_s[(size_t)n * C0N + tid];
  if (tid >= 64 && tid < 64 + C1N * 3) xv[tid - 64] = x_v[(size_t)n * C1N * 3 + tid - 64];
  __syncthreads();
  const float s2 = 0.70710678118654752f;
  const float s6 = 0.40824829046386302f;   // 1/sqrt(6)
  for (int c = tid; c < A1N; c += 256) {
    int ab = tA[c], a = ab >> 8, b = ab & 255;
    float ax = xv[a*3], ay = xv[a*3+1], az = xv[a*3+2];
    float bx = xv[b*3], by = xv[b*3+1], bz = xv[b*3+2];
    sve[c*3+0] = (ay*bz - az*by) * s2;
    sve[c*3+1] = (az*bx - ax*bz) * s2;
    sve[c*3+2] = (ax*by - ay*bx) * s2;
  }
  for (int q = tid; q < P1N; q += 256) {
    int ab = t1[q], a = ab >> 8, b = ab & 255;
    float cc = (a == b) ? s2 : 1.0f;
    float a0 = xv[a*3], a1 = xv[a*3+1], a2 = xv[a*3+2];
    float b0 = xv[b*3], b1 = xv[b*3+1], b2 = xv[b*3+2];
    sv2[q*5+0] = s2 * (a0*b1 + a1*b0) * cc;
    sv2[q*5+1] = s2 * (a1*b2 + a2*b1) * cc;
    sv2[q*5+2] = s2 * (a0*b2 + a2*b0) * cc;
    sv2[q*5+3] = s2 * (a0*b0 - a1*b1) * cc;
    sv2[q*5+4] = s6 * (2.0f*a2*b2 - a0*b0 - a1*b1) * cc;
  }
  __syncthreads();

  const u16* g  = gb + (size_t)n * GP;
  const u16* g1  = g + M0N;                 // 2672..4751
  const u16* g1e = g + M0N + M1N;           // 4752..5247
  const u16* g2  = g + M0N + M1N + A1N;     // 5248..5775
  float* o = out + (size_t)n * OSTR;

  if (tid < 96) {                                   // o1: (32 outs) x (3 comps)
    int oo = tid / 3, ii = tid - oo * 3;
    float acc = 0.0f;
    for (int c = 0; c < M1N; ++c) {
      float fc = (c < C1N) ? xv[c*3 + ii]
                           : xs[(c - C1N) >> 5] * xv[((c - C1N) & 31) * 3 + ii];
      acc += fc * bfu2f(g1[c]) * w1o[(size_t)c * 32 + oo];
    }
    o[64 + oo*3 + ii] = acc;
  } else if (tid < 144) {                           // o1e: 16 x 3
    int u = tid - 96, oo = u / 3, ii = u - oo * 3;
    float acc = 0.0f;
    for (int c = 0; c < A1N; ++c)
      acc += sve[c*3 + ii] * bfu2f(g1e[c]) * w1e[(size_t)c * 16 + oo];
    o[160 + oo*3 + ii] = acc;
  } else if (tid < 224) {                           // o2: 16 x 5
    int u = tid - 144, oo = u / 5, kk = u - oo * 5;
    float acc = 0.0f;
    for (int c = 0; c < P1N; ++c)
      acc += sv2[c*5 + kk] * bfu2f(g2[c]) * w2e[(size_t)c * 16 + oo];
    o[208 + oo*5 + kk] = acc;
  } else {                                          // copy o0 (64 f32)
    int u = tid - 224;
    o[u]      = o0p[(size_t)n * 128 + u];
    o[u + 32] = o0p[(size_t)n * 128 + u + 32];
  }
}

// ---------------- host launcher ----------------
extern "C" void kernel_launch(void* const* d_in, const int* in_sizes, int n_in,
                              void* d_out, int out_size, void* d_ws, size_t ws_size,
                              hipStream_t stream) {
  const float* x_s    = (const float*)d_in[0];
  const float* x_v    = (const float*)d_in[1];
  const float* mlp_w1 = (const float*)d_in[2];
  const float* mlp_w2 = (const float*)d_in[3];
  const float* w0     = (const float*)d_in[4];
  const float* w1o    = (const float*)d_in[5];
  const float* w1e    = (const float*)d_in[6];
  const float* w2e    = (const float*)d_in[7];
  float* out = (float*)d_out;

  char* base = (char*)d_ws;
  size_t off = 0;
  auto alloc = [&](size_t bytes) -> char* {
    char* p = base + off;
    off = (off + bytes + 255) & ~(size_t)255;
    return p;
  };
  int* t0   = (int*)alloc(P0N * 4);
  int* t1   = (int*)alloc(P1N * 4);
  int* tA   = (int*)alloc(A1N * 4);
  u16* w1b  = (u16*)alloc((size_t)KP * KP  * 2);
  u16* w2b  = (u16*)alloc((size_t)KP * GP  * 2);
  u16* w0b  = (u16*)alloc((size_t)KP * 128 * 2);
  u16* f0b  = (u16*)alloc((size_t)NSAMP * KP * 2);
  u16* hb   = (u16*)alloc((size_t)NSAMP * KP * 2);
  u16* gb   = (u16*)alloc((size_t)NSAMP * GP * 2);
  float* o0p = (float*)alloc((size_t)NSAMP * 128 * 4);
  if (ws_size < off) return;   // workspace too small: bail (nothing launched)

  init_tables<<<1, 4, 0, stream>>>(t0, t1, tA);

  { long tot = (long)KP * KP;  cvt_pad<<<(unsigned)((tot + 255) / 256), 256, 0, stream>>>(mlp_w1, w1b, M0N, M0N,  KP,  tot); }
  { long tot = (long)KP * GP;  cvt_pad<<<(unsigned)((tot + 255) / 256), 256, 0, stream>>>(mlp_w2, w2b, M0N, NIRR, GP,  tot); }
  { long tot = (long)KP * 128; cvt_pad<<<(unsigned)((tot + 255) / 256), 256, 0, stream>>>(w0,     w0b, M0N, 64,   128, tot); }

  build_f0<<<NSAMP, 256, 0, stream>>>(x_s, x_v, t0, t1, f0b);

  // h = silu(f0 @ W1)  -> bf16
  gemm_bf16_wmma<1><<<dim3(KP / 128, NSAMP / 128), 256, 0, stream>>>(f0b, w1b, hb, KP, KP, KP);
  // g = h @ W2  -> bf16
  gemm_bf16_wmma<2><<<dim3(GP / 128, NSAMP / 128), 256, 0, stream>>>(hb, w2b, gb, KP, GP, GP);
  // f0 *= g0 (in place)
  { long tot = (long)NSAMP * KP; gate_f0<<<(unsigned)((tot + 255) / 256), 256, 0, stream>>>(f0b, gb); }
  // o0 = (f0*g0) @ w0 -> f32 (padded to 128 cols)
  gemm_bf16_wmma<0><<<dim3(1, NSAMP / 128), 256, 0, stream>>>(f0b, w0b, o0p, KP, 128, 128);

  outputs_kernel<<<NSAMP, 256, 0, stream>>>(x_s, x_v, gb, o0p, w1o, w1e, w2e, tA, t1, out);
}
